// DiffusionAttentionPairBias_29523605192848
// MI455X (gfx1250) — matrix operand order repaired and checked by reading, not compiled, not run on hardware
//
#include <hip/hip_runtime.h>
#include <math.h>

// ---------------------------------------------------------------------------
// CDNA5 (gfx1250) wave32 WMMA implementation of DiffusionAttentionPairBias.
// All matrix math in bf16 WMMA (v_wmma_f32_16x16x32_bf16), f32 accumulate.
// z-stream (603 MB) staged into LDS via TDM tensor_load_to_lds.
// ---------------------------------------------------------------------------

typedef __bf16 bf16;
typedef __attribute__((ext_vector_type(16))) __bf16 v16bf;
typedef __attribute__((ext_vector_type(8)))  __bf16 v8bf;
typedef __attribute__((ext_vector_type(8)))  float  v8f;
typedef __attribute__((ext_vector_type(4)))  unsigned int u32x4;
typedef __attribute__((ext_vector_type(8)))  unsigned int u32x8;

static __device__ __forceinline__ v8f wmma_bf16(v16bf a, v16bf b, v8f c) {
  // 8 args: (neg_a, A, neg_b, B, c_mod, C, reuse_a, reuse_b)
  return __builtin_amdgcn_wmma_f32_16x16x32_bf16(false, a, false, b, (short)0, c,
                                                 false, false);
}

// Assemble a 16-element bf16 fragment from two 16-byte-aligned halves.
static __device__ __forceinline__ v16bf load_frag(const bf16* lo, const bf16* hi) {
  union { v16bf v; v8bf h[2]; } u;
  u.h[0] = *reinterpret_cast<const v8bf*>(lo);
  u.h[1] = *reinterpret_cast<const v8bf*>(hi);
  return u.v;
}

static __device__ __forceinline__ float sigmoidf_(float x) {
  return 1.0f / (1.0f + __expf(-x));
}

// --------------------------- elementwise kernels ---------------------------

// Wt[n*K + k] = bf16(W[k*N + n])   (transpose + convert, weights)
__global__ void k_wt(const float* __restrict__ W, bf16* __restrict__ Wt, int K, int N) {
  int i = blockIdx.x * blockDim.x + threadIdx.x;
  if (i >= K * N) return;
  int k = i % K, n = i / K;
  Wt[i] = (bf16)W[(size_t)k * N + n];
}

__global__ void k_cvt(const float* __restrict__ x, bf16* __restrict__ y, int n) {
  int i = blockIdx.x * blockDim.x + threadIdx.x;
  if (i < n) y[i] = (bf16)x[i];
}

// Row LayerNorm (no affine). Optional f32 and/or bf16 outputs.
__global__ void k_ln(const float* __restrict__ x, float* __restrict__ yf,
                     bf16* __restrict__ yb, int C) {
  __shared__ float red[256];
  int row = blockIdx.x, t = threadIdx.x;
  const float* xr = x + (size_t)row * C;
  float s = 0.f;
  for (int c = t; c < C; c += 256) s += xr[c];
  red[t] = s; __syncthreads();
  for (int o = 128; o > 0; o >>= 1) { if (t < o) red[t] += red[t + o]; __syncthreads(); }
  float mean = red[0] / (float)C;
  __syncthreads();
  float v = 0.f;
  for (int c = t; c < C; c += 256) { float d = xr[c] - mean; v += d * d; }
  red[t] = v; __syncthreads();
  for (int o = 128; o > 0; o >>= 1) { if (t < o) red[t] += red[t + o]; __syncthreads(); }
  float rstd = rsqrtf(red[0] / (float)C + 1e-5f);
  for (int c = t; c < C; c += 256) {
    float val = (xr[c] - mean) * rstd;
    if (yf) yf[(size_t)row * C + c] = val;
    if (yb) yb[(size_t)row * C + c] = (bf16)val;
  }
}

// a_norm = sigmoid(G) * a_ln + Bd   (stored bf16 for the QKV GEMMs)
__global__ void k_combine(const float* __restrict__ G, const float* __restrict__ Bd,
                          const float* __restrict__ a_ln, bf16* __restrict__ out, int n) {
  int i = blockIdx.x * blockDim.x + threadIdx.x;
  if (i < n) out[i] = (bf16)(sigmoidf_(G[i]) * a_ln[i] + Bd[i]);
}

// [B,N, h*48+d] f32  ->  [B,H,N,64] bf16 (d>=48 zero-padded)
__global__ void k_pack_qk(const float* __restrict__ src, bf16* __restrict__ dst,
                          int Bc, int Hc, int Nc, int CT) {
  int i = blockIdx.x * blockDim.x + threadIdx.x;
  if (i >= Bc * Hc * Nc * 64) return;
  int d = i & 63;
  int n = (i >> 6) % Nc;
  int bh = (i >> 6) / Nc;
  int h = bh % Hc, b = bh / Hc;
  float v = (d < 48) ? src[(size_t)(b * Nc + n) * CT + h * 48 + d] : 0.f;
  dst[i] = (bf16)v;
}

// [B,N, h*48+d] f32  ->  [B,H,64,N] bf16 (transposed V, zero pad)
__global__ void k_pack_vt(const float* __restrict__ src, bf16* __restrict__ dst,
                          int Bc, int Hc, int Nc, int CT) {
  int i = blockIdx.x * blockDim.x + threadIdx.x;
  if (i >= Bc * Hc * 64 * Nc) return;
  int n = i % Nc;
  int d = (i / Nc) & 63;
  int bh = i / (Nc * 64);
  int h = bh % Hc, b = bh / Hc;
  float v = (d < 48) ? src[(size_t)(b * Nc + n) * CT + h * 48 + d] : 0.f;
  dst[i] = (bf16)v;
}

// out = sigmoid(og) * out1
__global__ void k_final(const float* __restrict__ out1, const float* __restrict__ og,
                        float* __restrict__ out, int n) {
  int i = blockIdx.x * blockDim.x + threadIdx.x;
  if (i < n) out[i] = sigmoidf_(og[i]) * out1[i];
}

// ------------------------------ WMMA GEMM ----------------------------------
// C[M,Nt] = A[M,K](bf16) * Bt[Nt,K](bf16, pre-transposed) + bias[Nt]?
// One wave computes a 16x64 tile (A-fragment reused across 4 WMMAs per K-step).
__global__ void k_gemm(const bf16* __restrict__ A, const bf16* __restrict__ Bt,
                       const float* __restrict__ bias, float* __restrict__ C,
                       int M, int Nt, int K) {
  int tiles_n = Nt >> 6;                 // 64-wide N tiles
  int tile = blockIdx.x;
  int tm = (tile / tiles_n) << 4, tn = (tile % tiles_n) << 6;
  int l = threadIdx.x, lm = l & 15, lh = l >> 4;
  const bf16* arow = A + (size_t)(tm + lm) * K + lh * 8;   // A-frag: row lm, K-half lh
  const bf16* brow = Bt + (size_t)(tn + lm) * K + lh * 8;  // B-frag: col lm, K-half lh
  const size_t bstr = (size_t)16 * K;
  v8f acc[4] = {};
  for (int k0 = 0; k0 < K; k0 += 32) {
    v16bf av = load_frag(arow + k0, arow + k0 + 16);
#pragma unroll
    for (int j = 0; j < 4; ++j) {
      const bf16* bp = brow + (size_t)j * bstr + k0;
      acc[j] = wmma_bf16(av, load_frag(bp, bp + 16), acc[j]);
    }
  }
#pragma unroll
  for (int j = 0; j < 4; ++j) {
    int col = tn + j * 16 + lm;
    float bsv = bias ? bias[col] : 0.f;
#pragma unroll
    for (int r = 0; r < 8; ++r) {
      int row = tm + r + lh * 8;
      C[(size_t)row * Nt + col] = acc[j][r] + bsv;
    }
  }
}

// -------------------- fused LN(z)*gamma+beta @ Wpb + mask -------------------
// One wave per 16 flattened (b,i,j) rows of 128 channels.
// The 8 KB z tile is DMA'd into LDS by the Tensor Data Mover.
__global__ void k_pairbias(const float* __restrict__ z, const float* __restrict__ mask,
                           const bf16* __restrict__ WpbT,   // [H=16][128]
                           const float* __restrict__ gamma, const float* __restrict__ beta,
                           float* __restrict__ pb,          // [B,H,N,N]
                           int Nc, int Hc) {
  __shared__ float Z[16][128];
  __shared__ float mean_s[16], rstd_s[16];
  int l = threadIdx.x, lm = l & 15, lh = l >> 4;
  size_t p0 = (size_t)blockIdx.x * 16;

  // ---- TDM: 1-D tile, 2048 x 4B elements, contiguous -> LDS ----
  unsigned lds_off = (unsigned)(uintptr_t)(&Z[0][0]);
  unsigned long long ga = (unsigned long long)(uintptr_t)(z + p0 * 128);
  u32x4 g0;
  g0[0] = 1u;                                           // count=1, user mode
  g0[1] = lds_off;                                      // lds_addr
  g0[2] = (unsigned)(ga & 0xFFFFFFFFu);                 // global_addr[31:0]
  g0[3] = (unsigned)((ga >> 32) & 0x01FFFFFFu) | 0x80000000u; // addr[56:32]|type=2
  u32x8 g1;
  g1[0] = 0x00020000u;   // workgroup_mask=0, data_size=2 (4B)
  g1[1] = 2048u << 16;   // tensor_dim0[15:0] in [31:16]
  g1[2] = 1u << 16;      // tensor_dim0[31:16]=0, tensor_dim1[15:0]=1
  g1[3] = 2048u << 16;   // tile_dim0 = 2048
  g1[4] = 0u;            // tile_dim1 = tile_dim2 = 0 (unused)
  g1[5] = 2048u;         // tensor_dim0_stride[31:0]
  g1[6] = 0u;
  g1[7] = 0u;
  asm volatile("tensor_load_to_lds %0, %1" :: "s"(g0), "s"(g1) : "memory");
  __builtin_amdgcn_s_wait_tensorcnt(0);
  __syncthreads();

  if (l < 16) {
    float m = 0.f;
#pragma unroll
    for (int c = 0; c < 128; ++c) m += Z[l][c];
    m *= (1.0f / 128.0f);
    float v = 0.f;
#pragma unroll
    for (int c = 0; c < 128; ++c) { float d = Z[l][c] - m; v += d * d; }
    mean_s[l] = m;
    rstd_s[l] = rsqrtf(v * (1.0f / 128.0f) + 1e-5f);
  }
  __syncthreads();
  float mu = mean_s[lm], rs = rstd_s[lm];
  v8f acc = {};
#pragma unroll
  for (int kk = 0; kk < 4; ++kk) {
    v16bf av;
#pragma unroll
    for (int e = 0; e < 16; ++e) {
      int k = kk * 32 + ((e < 8) ? (lh * 8 + e) : (16 + lh * 8 + (e - 8)));
      av[e] = (bf16)((Z[lm][k] - mu) * rs * gamma[k] + beta[k]);
    }
    const bf16* wp = WpbT + lm * 128 + kk * 32 + lh * 8;
    v16bf bv = load_frag(wp, wp + 16);
    acc = wmma_bf16(av, bv, acc);
  }
#pragma unroll
  for (int r = 0; r < 8; ++r) {
    size_t p = p0 + r + lh * 8;          // flat (b,i,j)
    int j = (int)(p % Nc);
    size_t bi = p / Nc;
    int i2 = (int)(bi % Nc);
    int b = (int)(bi / Nc);
    pb[(((size_t)(b * Hc + lm)) * Nc + i2) * Nc + j] = acc[r] + mask[p];
  }
}

// ---------------------- flash attention (one wave/q-tile) -------------------
__global__ void k_attn(const bf16* __restrict__ Qp,   // [B,H,N,64]
                       const bf16* __restrict__ Kp,   // [B,H,N,64]
                       const bf16* __restrict__ Vt,   // [B,H,64,N]
                       const float* __restrict__ pb,  // [B,H,N,N]
                       const float* __restrict__ gatef, // [B,N,768]
                       bf16* __restrict__ Obf,        // [B,N,768]
                       int Bc, int Hc, int Nc, int CT) {
  __shared__ float Ls[16][32];
  __shared__ bf16  Ps[16][32];
  __shared__ float ms[16], ss[16], als[16];
  int l = threadIdx.x, lm = l & 15, lh = l >> 4;
  int qtiles = Nc >> 4;
  int qt = blockIdx.x % qtiles;
  int bh = blockIdx.x / qtiles;           // b*H + h
  int h = bh % Hc, b = bh / Hc;
  const float scale = 0.14433756729740643f;  // 1/sqrt(48)
  size_t baseQ  = ((size_t)bh * Nc + qt * 16) * 64;
  size_t baseK  = (size_t)bh * Nc * 64;
  size_t baseV  = (size_t)bh * 64 * Nc;
  size_t basePB = ((size_t)bh * Nc + qt * 16) * Nc;

  const bf16* qrow = Qp + baseQ + (size_t)lm * 64 + lh * 16;
  v16bf aq0 = load_frag(qrow, qrow + 8);         // d = 0..31
  v16bf aq1 = load_frag(qrow + 32, qrow + 40);   // d = 32..63
  if (l < 16) { ms[l] = -1e30f; ss[l] = 0.f; }
  v8f O0 = {}, O1 = {}, O2 = {}, O3 = {};
  __syncthreads();

  for (int k0 = 0; k0 < Nc; k0 += 32) {
#pragma unroll
    for (int sub = 0; sub < 2; ++sub) {
      int kt = k0 + sub * 16;
      const bf16* krow = Kp + baseK + (size_t)(kt + lm) * 64 + lh * 16;
      v16bf bk0 = load_frag(krow, krow + 8);        // d = 0..31
      v16bf bk1 = load_frag(krow + 32, krow + 40);  // d = 32..63
      v8f cl = {};
      cl = wmma_bf16(aq0, bk0, cl);
      cl = wmma_bf16(aq1, bk1, cl);
#pragma unroll
      for (int r = 0; r < 8; ++r) {
        int row = r + lh * 8;
        Ls[row][sub * 16 + lm] =
            cl[r] * scale + pb[basePB + (size_t)row * Nc + kt + lm];
      }
    }
    __syncthreads();
    if (l < 16) {  // online softmax stats per q-row
      float mo = ms[l], tm = mo;
#pragma unroll
      for (int c = 0; c < 32; ++c) tm = fmaxf(tm, Ls[l][c]);
      float alpha = __expf(mo - tm);
      float ts = 0.f;
#pragma unroll
      for (int c = 0; c < 32; ++c) {
        float p = __expf(Ls[l][c] - tm);
        Ps[l][c] = (bf16)p;
        ts += p;
      }
      ss[l] = ss[l] * alpha + ts;
      ms[l] = tm;
      als[l] = alpha;
    }
    __syncthreads();
    // P fragment (16x32) from LDS in A-layout
    v16bf ap;
#pragma unroll
    for (int e = 0; e < 16; ++e) {
      int kk = lh * 16 + e;
      ap[e] = Ps[lm][kk];
    }
    float al[8];
#pragma unroll
    for (int r = 0; r < 8; ++r) al[r] = als[r + lh * 8];
#pragma unroll
    for (int r = 0; r < 8; ++r) {
      O0[r] *= al[r]; O1[r] *= al[r]; O2[r] *= al[r]; O3[r] *= al[r];
    }
    const bf16* vb = Vt + baseV + k0 + lh * 16;
    const bf16* v0 = vb + (size_t)(0 + lm) * Nc;
    O0 = wmma_bf16(ap, load_frag(v0, v0 + 8), O0);
    const bf16* v1 = vb + (size_t)(16 + lm) * Nc;
    O1 = wmma_bf16(ap, load_frag(v1, v1 + 8), O1);
    const bf16* v2 = vb + (size_t)(32 + lm) * Nc;
    O2 = wmma_bf16(ap, load_frag(v2, v2 + 8), O2);
    const bf16* v3 = vb + (size_t)(48 + lm) * Nc;
    O3 = wmma_bf16(ap, load_frag(v3, v3 + 8), O3);
    __syncthreads();
  }
  // gated epilogue: o = sigmoid(gate) * (O / sum)
#pragma unroll
  for (int r = 0; r < 8; ++r) {
    int row = r + lh * 8;
    float inv = 1.0f / ss[row];
    int q = qt * 16 + row;
    size_t obase = (size_t)(b * Nc + q) * CT + h * 48;
    float vals[4] = { O0[r], O1[r], O2[r], O3[r] };
#pragma unroll
    for (int j = 0; j < 4; ++j) {
      int d = j * 16 + lm;
      if (d < 48) {
        float g = sigmoidf_(gatef[obase + d]);
        Obf[obase + d] = (bf16)(g * vals[j] * inv);
      }
    }
  }
}

// ------------------------------- launcher ----------------------------------
extern "C" void kernel_launch(void* const* d_in, const int* in_sizes, int n_in,
                              void* d_out, int out_size, void* d_ws, size_t ws_size,
                              hipStream_t stream) {
  (void)in_sizes; (void)n_in; (void)out_size; (void)ws_size;
  const int B = 2, N = 768, CT = 768, CS = 384, H = 16, CP = 128;
  const int BN = B * N;

  const float* a     = (const float*)d_in[0];
  const float* s     = (const float*)d_in[1];
  const float* z     = (const float*)d_in[2];
  const float* mask  = (const float*)d_in[3];
  const float* Wg    = (const float*)d_in[4];
  const float* bg    = (const float*)d_in[5];
  const float* Wb    = (const float*)d_in[6];
  const float* Wq    = (const float*)d_in[7];
  const float* bq    = (const float*)d_in[8];
  const float* Wk    = (const float*)d_in[9];
  const float* Wv    = (const float*)d_in[10];
  const float* Wpb   = (const float*)d_in[11];
  const float* gam   = (const float*)d_in[12];
  const float* bet   = (const float*)d_in[13];
  const float* Wgate = (const float*)d_in[14];
  const float* Wout  = (const float*)d_in[15];
  const float* Wog   = (const float*)d_in[16];
  const float* bog   = (const float*)d_in[17];
  float* out = (float*)d_out;

  char* w = (char*)d_ws;
  size_t off = 0;
  auto alloc = [&](size_t bytes) -> void* {
    void* p = w + off;
    off += (bytes + 255) & ~(size_t)255;
    return p;
  };
  float* a_ln = (float*)alloc((size_t)BN * CT * 4);
  float* fA   = (float*)alloc((size_t)BN * CT * 4);   // G -> qf -> out1
  float* fB   = (float*)alloc((size_t)BN * CT * 4);   // Bd -> kf -> og
  float* fC   = (float*)alloc((size_t)BN * CT * 4);   // vf
  float* fD   = (float*)alloc((size_t)BN * CT * 4);   // gatef
  float* pair = (float*)alloc((size_t)B * H * N * N * 4);
  bf16* s_ln_bf = (bf16*)alloc((size_t)BN * CS * 2);
  bf16* s_bf    = (bf16*)alloc((size_t)BN * CS * 2);
  bf16* anorm   = (bf16*)alloc((size_t)BN * CT * 2);
  bf16* o_bf    = (bf16*)alloc((size_t)BN * CT * 2);
  bf16* qp      = (bf16*)alloc((size_t)B * H * N * 64 * 2);
  bf16* kp      = (bf16*)alloc((size_t)B * H * N * 64 * 2);
  bf16* vt      = (bf16*)alloc((size_t)B * H * 64 * N * 2);
  bf16* WgT    = (bf16*)alloc((size_t)CS * CT * 2);
  bf16* WbT    = (bf16*)alloc((size_t)CS * CT * 2);
  bf16* WqT    = (bf16*)alloc((size_t)CT * CT * 2);
  bf16* WkT    = (bf16*)alloc((size_t)CT * CT * 2);
  bf16* WvT    = (bf16*)alloc((size_t)CT * CT * 2);
  bf16* WgateT = (bf16*)alloc((size_t)CT * CT * 2);
  bf16* WoutT  = (bf16*)alloc((size_t)CT * CT * 2);
  bf16* WogT   = (bf16*)alloc((size_t)CS * CT * 2);
  bf16* WpbT   = (bf16*)alloc((size_t)CP * H * 2);

  auto cdiv = [](int x, int y) { return (x + y - 1) / y; };

  // Weight transposes/conversions
  k_wt<<<cdiv(CS * CT, 256), 256, 0, stream>>>(Wg, WgT, CS, CT);
  k_wt<<<cdiv(CS * CT, 256), 256, 0, stream>>>(Wb, WbT, CS, CT);
  k_wt<<<cdiv(CT * CT, 256), 256, 0, stream>>>(Wq, WqT, CT, CT);
  k_wt<<<cdiv(CT * CT, 256), 256, 0, stream>>>(Wk, WkT, CT, CT);
  k_wt<<<cdiv(CT * CT, 256), 256, 0, stream>>>(Wv, WvT, CT, CT);
  k_wt<<<cdiv(CT * CT, 256), 256, 0, stream>>>(Wgate, WgateT, CT, CT);
  k_wt<<<cdiv(CT * CT, 256), 256, 0, stream>>>(Wout, WoutT, CT, CT);
  k_wt<<<cdiv(CS * CT, 256), 256, 0, stream>>>(Wog, WogT, CS, CT);
  k_wt<<<cdiv(CP * H, 256), 256, 0, stream>>>(Wpb, WpbT, CP, H);

  // LayerNorms + raw s conversion
  k_ln<<<BN, 256, 0, stream>>>(a, a_ln, (bf16*)nullptr, CT);
  k_ln<<<BN, 256, 0, stream>>>(s, (float*)nullptr, s_ln_bf, CS);
  k_cvt<<<cdiv(BN * CS, 256), 256, 0, stream>>>(s, s_bf, BN * CS);

  const int g768 = (BN / 16) * (CT / 64);   // 16x64 tiles per wave
  // AdaLN conditioning projections
  k_gemm<<<g768, 32, 0, stream>>>(s_ln_bf, WgT, bg, fA, BN, CT, CS);
  k_gemm<<<g768, 32, 0, stream>>>(s_ln_bf, WbT, (const float*)nullptr, fB, BN, CT, CS);
  k_combine<<<cdiv(BN * CT, 256), 256, 0, stream>>>(fA, fB, a_ln, anorm, BN * CT);

  // QKV + gate projections
  k_gemm<<<g768, 32, 0, stream>>>(anorm, WqT, bq, fA, BN, CT, CT);
  k_gemm<<<g768, 32, 0, stream>>>(anorm, WkT, (const float*)nullptr, fB, BN, CT, CT);
  k_gemm<<<g768, 32, 0, stream>>>(anorm, WvT, (const float*)nullptr, fC, BN, CT, CT);
  k_gemm<<<g768, 32, 0, stream>>>(anorm, WgateT, (const float*)nullptr, fD, BN, CT, CT);
  k_pack_qk<<<cdiv(B * H * N * 64, 256), 256, 0, stream>>>(fA, qp, B, H, N, CT);
  k_pack_qk<<<cdiv(B * H * N * 64, 256), 256, 0, stream>>>(fB, kp, B, H, N, CT);
  k_pack_vt<<<cdiv(B * H * 64 * N, 256), 256, 0, stream>>>(fC, vt, B, H, N, CT);

  // Pair bias: single pass over 603 MB z (TDM -> LDS), fused LN + head proj + mask
  k_pairbias<<<(B * N * N) / 16, 32, 0, stream>>>(z, mask, WpbT, gam, bet, pair, N, H);

  // Flash attention with pair bias + gating
  k_attn<<<B * H * (N / 16), 32, 0, stream>>>(qp, kp, vt, pair, fD, o_bf, B, H, N, CT);

  // Output projection + adaLN-Zero gate
  k_gemm<<<g768, 32, 0, stream>>>(o_bf, WoutT, (const float*)nullptr, fA, BN, CT, CT);
  k_gemm<<<g768, 32, 0, stream>>>(s_bf, WogT, bog, fB, BN, CT, CS);
  k_final<<<cdiv(BN * CT, 256), 256, 0, stream>>>(fA, fB, out, BN * CT);
}
